// GIN_44324062494962
// MI455X (gfx1250) — matrix operand order repaired and checked by reading, not compiled, not run on hardware
//
#include <hip/hip_runtime.h>
#include <hip/hip_bf16.h>
#include <math.h>

// ---------------------------------------------------------------------------
// GIN 2-layer GNN for MI455X (gfx1250, wave32, WMMA)
//   layer: h = relu( relu((x + scatter_sum(x)) @ Wa + ba) @ Wb + bb )
//   final: log_softmax over 64 channels
// ---------------------------------------------------------------------------

typedef __attribute__((ext_vector_type(16))) __bf16 v16bf;
typedef __attribute__((ext_vector_type(8)))  float  v8f;

#define GNN_NODES 50000
#define GNN_EDGES 800000
#define GNN_INCH  128
#define GNN_HID   64
#define GNN_TILES (GNN_NODES / 16)   // 3125, exact

// ---------------------------------------------------------------------------
// float4 grid-stride copy (h0 = x  /  h2in = h1 : the "(1+eps)*x" term, eps=0)
// ---------------------------------------------------------------------------
__global__ void gin_copy_f4(const float4* __restrict__ s, float4* __restrict__ d, int n4) {
    int i = blockIdx.x * blockDim.x + threadIdx.x;
    int stride = gridDim.x * blockDim.x;
    for (; i < n4; i += stride) d[i] = s[i];
}

// ---------------------------------------------------------------------------
// Transpose + convert weights to bf16, [K][N] fp32 -> [N][K] bf16 so that a
// WMMA B-fragment (one column, 16 consecutive K) is one contiguous 32B load.
// ---------------------------------------------------------------------------
__global__ void gin_prep_weights(const float* __restrict__ w1a, const float* __restrict__ w1b,
                                 const float* __restrict__ w2a, const float* __restrict__ w2b,
                                 __bf16* __restrict__ w1aT, __bf16* __restrict__ w1bT,
                                 __bf16* __restrict__ w2aT, __bf16* __restrict__ w2bT) {
    int i = blockIdx.x * blockDim.x + threadIdx.x;
    if (i < GNN_INCH * GNN_HID) {
        int k = i / GNN_HID, n = i % GNN_HID;
        w1aT[n * GNN_INCH + k] = (__bf16)w1a[i];
    }
    if (i < GNN_HID * GNN_HID) {
        int k = i / GNN_HID, n = i % GNN_HID;
        w1bT[n * GNN_HID + k] = (__bf16)w1b[i];
        w2aT[n * GNN_HID + k] = (__bf16)w2a[i];
        w2bT[n * GNN_HID + k] = (__bf16)w2b[i];
    }
}

// ---------------------------------------------------------------------------
// Edge scatter-add: one wave per edge. Feature rows are L2-resident (25.6 MB
// total << 192 MB L2), so this is an L2 atomic-throughput kernel.
// CH=128: one float4 per lane; CH=64: one float2 per lane.
// ---------------------------------------------------------------------------
template<int CH>
__global__ __launch_bounds__(256) void gin_scatter_add(const float* __restrict__ feat,
                                                       const int* __restrict__ srcIdx,
                                                       const int* __restrict__ dstIdx,
                                                       float* accum, int nEdges) {
    int wid  = (blockIdx.x * blockDim.x + threadIdx.x) >> 5;
    int lane = threadIdx.x & 31;
    if (wid >= nEdges) return;
    unsigned s = (unsigned)srcIdx[wid];
    unsigned d = (unsigned)dstIdx[wid];
    if (s >= (unsigned)GNN_NODES || d >= (unsigned)GNN_NODES) return;
    const float* fp = feat + (size_t)s * CH;
    float* ap = accum + (size_t)d * CH;
    if (CH == 128) {
        float4 v = *(const float4*)(fp + lane * 4);
        unsafeAtomicAdd(ap + lane * 4 + 0, v.x);
        unsafeAtomicAdd(ap + lane * 4 + 1, v.y);
        unsafeAtomicAdd(ap + lane * 4 + 2, v.z);
        unsafeAtomicAdd(ap + lane * 4 + 3, v.w);
    } else {
        float2 v = *(const float2*)(fp + lane * 2);
        unsafeAtomicAdd(ap + lane * 2 + 0, v.x);
        unsafeAtomicAdd(ap + lane * 2 + 1, v.y);
    }
}

// ---------------------------------------------------------------------------
// Fused 2-linear MLP per 16-node tile, one wave per tile (8 waves / block).
//   midT = relu(in[16xK] @ WaT' + ba)          (GEMM-A, bf16 WMMA, f32 acc)
//   outT = mid @ WbT' + bb                      (GEMM-B)
//   FINAL=false: relu -> out fp32; FINAL=true: row log_softmax -> out fp32
//
// WMMA 16x16x32 bf16 fragment layouts (wave32):
//   A: lane%16 = row M; K = ks + (lane/16)*16 + e (same order as B)
//   B: col N = lane%16; K = ks + (lane/16)*16 + e   (contiguous in WT[N][K])
//   C/D: row = r + 8*(lane/16), col = lane%16 per v8f element r
// ---------------------------------------------------------------------------
template<int KDIM, bool FINAL>
__global__ __launch_bounds__(256) void gin_mlp(const float* __restrict__ in,    // [N][KDIM]
                                               const __bf16* __restrict__ wAT,  // [64][KDIM]
                                               const float* __restrict__ bA,    // [64]
                                               const __bf16* __restrict__ wBT,  // [64][64]
                                               const float* __restrict__ bB,    // [64]
                                               float* __restrict__ out) {       // [N][64]
    __shared__ __bf16 smid[8 * 16 * GNN_HID];                  // 16 KB
    __shared__ float  sout[FINAL ? 8 * 16 * GNN_HID : 1];      // 32 KB (final only)

    const int wv   = threadIdx.x >> 5;
    const int lane = threadIdx.x & 31;
    const int tile = blockIdx.x * 8 + wv;
    if (tile >= GNN_TILES) return;                 // wave-uniform: EXEC stays full
    const int row  = lane & 15;
    const int half = lane >> 4;
    const int h8   = half * 8;
    const int h16  = half * 16;

    const v8f vzero = {0.f, 0.f, 0.f, 0.f, 0.f, 0.f, 0.f, 0.f};

    // ---------------- GEMM A : in(16 x KDIM) @ Wa(KDIM x 64) ----------------
    v8f cA[4];
    #pragma unroll
    for (int t = 0; t < 4; ++t) cA[t] = vzero;

    const float* arow = in + (size_t)(tile * 16 + row) * KDIM;
    #pragma unroll
    for (int ks = 0; ks < KDIM; ks += 32) {
        v16bf a;
        #pragma unroll
        for (int e = 0; e < 8; ++e) a[e]     = (__bf16)arow[ks + h16 + e];
        #pragma unroll
        for (int e = 0; e < 8; ++e) a[8 + e] = (__bf16)arow[ks + h16 + 8 + e];
        #pragma unroll
        for (int t = 0; t < 4; ++t) {
            v16bf b = *(const v16bf*)(wAT + (size_t)(t * 16 + row) * KDIM + ks + h16);
            cA[t] = __builtin_amdgcn_wmma_f32_16x16x32_bf16(false, a, false, b,
                                                            (short)0, cA[t], false, false);
        }
    }

    // bias + relu, stage mid (16x64 bf16) in per-wave LDS for D->A relayout
    #pragma unroll
    for (int t = 0; t < 4; ++t) {
        float bias = bA[t * 16 + row];
        #pragma unroll
        for (int r = 0; r < 8; ++r) {
            float v = fmaxf(cA[t][r] + bias, 0.f);
            smid[wv * 16 * GNN_HID + (r + h8) * GNN_HID + t * 16 + row] = (__bf16)v;
        }
    }
    asm volatile("s_wait_dscnt 0" ::: "memory");   // within-wave LDS RAW

    // ---------------- GEMM B : mid(16 x 64) @ Wb(64 x 64) ----------------
    v8f cB[4];
    #pragma unroll
    for (int t = 0; t < 4; ++t) cB[t] = vzero;
    const __bf16* mrow = &smid[wv * 16 * GNN_HID + row * GNN_HID];
    #pragma unroll
    for (int ks = 0; ks < GNN_HID; ks += 32) {
        v16bf a;
        #pragma unroll
        for (int e = 0; e < 8; ++e) a[e]     = mrow[ks + h16 + e];
        #pragma unroll
        for (int e = 0; e < 8; ++e) a[8 + e] = mrow[ks + h16 + 8 + e];
        #pragma unroll
        for (int t = 0; t < 4; ++t) {
            v16bf b = *(const v16bf*)(wBT + (size_t)(t * 16 + row) * GNN_HID + ks + h16);
            cB[t] = __builtin_amdgcn_wmma_f32_16x16x32_bf16(false, a, false, b,
                                                            (short)0, cB[t], false, false);
        }
    }

    if (!FINAL) {
        // layer-1 epilogue: + bias, outer relu, store fp32
        #pragma unroll
        for (int t = 0; t < 4; ++t) {
            float bias = bB[t * 16 + row];
            #pragma unroll
            for (int r = 0; r < 8; ++r) {
                float v = fmaxf(cB[t][r] + bias, 0.f);
                out[(size_t)(tile * 16 + r + h8) * GNN_HID + t * 16 + row] = v;
            }
        }
    } else {
        // final epilogue: logits -> LDS, per-row log_softmax
        #pragma unroll
        for (int t = 0; t < 4; ++t) {
            float bias = bB[t * 16 + row];
            #pragma unroll
            for (int r = 0; r < 8; ++r) {
                sout[wv * 16 * GNN_HID + (r + h8) * GNN_HID + t * 16 + row] = cB[t][r] + bias;
            }
        }
        asm volatile("s_wait_dscnt 0" ::: "memory");
        if (lane < 16) {
            const float* rp = &sout[wv * 16 * GNN_HID + lane * GNN_HID];
            float m = -INFINITY;
            #pragma unroll
            for (int c = 0; c < GNN_HID; ++c) m = fmaxf(m, rp[c]);
            float ssum = 0.f;
            #pragma unroll
            for (int c = 0; c < GNN_HID; ++c) ssum += expf(rp[c] - m);
            float lse = m + logf(ssum);
            float* op = out + (size_t)(tile * 16 + lane) * GNN_HID;
            #pragma unroll
            for (int c = 0; c < GNN_HID; ++c) op[c] = rp[c] - lse;
        }
    }
}

// ---------------------------------------------------------------------------
// Launcher. Workspace layout (bytes):
//   [0,           25.6M)  h0   : x + scatter_sum(x)    (50000 x 128 f32)
//                         h2in : h1 + scatter_sum(h1)  (reuses h0, h0 dead)
//   [25.6M,       38.4M)  h1   : layer-1 output        (50000 x 64 f32)
//   [38.4M,  38.4M+40KB)  bf16 transposed weights
// ---------------------------------------------------------------------------
extern "C" void kernel_launch(void* const* d_in, const int* in_sizes, int n_in,
                              void* d_out, int out_size, void* d_ws, size_t ws_size,
                              hipStream_t stream) {
    const float* x   = (const float*)d_in[0];
    const int*   ei  = (const int*)d_in[1];
    const float* w1a = (const float*)d_in[2];
    const float* b1a = (const float*)d_in[3];
    const float* w1b = (const float*)d_in[4];
    const float* b1b = (const float*)d_in[5];
    const float* w2a = (const float*)d_in[6];
    const float* b2a = (const float*)d_in[7];
    const float* w2b = (const float*)d_in[8];
    const float* b2b = (const float*)d_in[9];

    const int* src = ei;
    const int* dst = ei + GNN_EDGES;

    char* ws = (char*)d_ws;
    float*  h0   = (float*)(ws);
    float*  h1   = (float*)(ws + 25600000);
    float*  h2in = (float*)(ws);                         // reuse h0 space
    __bf16* w1aT = (__bf16*)(ws + 38400000);             // 64 x 128
    __bf16* w1bT = (__bf16*)(ws + 38400000 + 16384);     // 64 x 64
    __bf16* w2aT = (__bf16*)(ws + 38400000 + 24576);     // 64 x 64
    __bf16* w2bT = (__bf16*)(ws + 38400000 + 32768);     // 64 x 64

    const int mlpBlocks = (GNN_TILES + 7) / 8;           // 391

    gin_prep_weights<<<(GNN_INCH * GNN_HID + 255) / 256, 256, 0, stream>>>(
        w1a, w1b, w2a, w2b, w1aT, w1bT, w2aT, w2bT);

    // ---- layer 1 ----
    gin_copy_f4<<<2048, 256, 0, stream>>>((const float4*)x, (float4*)h0,
                                          GNN_NODES * GNN_INCH / 4);
    gin_scatter_add<GNN_INCH><<<GNN_EDGES / 8, 256, 0, stream>>>(x, src, dst, h0, GNN_EDGES);
    gin_mlp<GNN_INCH, false><<<mlpBlocks, 256, 0, stream>>>(h0, w1aT, b1a, w1bT, b1b, h1);

    // ---- layer 2 + log_softmax ----
    gin_copy_f4<<<2048, 256, 0, stream>>>((const float4*)h1, (float4*)h2in,
                                          GNN_NODES * GNN_HID / 4);
    gin_scatter_add<GNN_HID><<<GNN_EDGES / 8, 256, 0, stream>>>(h1, src, dst, h2in, GNN_EDGES);
    gin_mlp<GNN_HID, true><<<mlpBlocks, 256, 0, stream>>>(h2in, w2aT, b2a, w2bT, b2b,
                                                          (float*)d_out);
}
